// MultiHeadAttention_63685775065336
// MI455X (gfx1250) — compile-verified, matrix operand order checked by reading, not run on hardware
//
#include <hip/hip_runtime.h>

#define S_LEN   2048
#define DMODEL  1024
#define NHEADS  16
#define DDEPTH  64
#define RKER    64
#define CHUNK   32
#define NORM_D  0.35355339059327379f  /* 64^(-1/4) */
#define RSQR_D  0.125f                /* 1/sqrt(64) */
#define EPSV    1e-6f

// gfx1250 async global->LDS copies (inline asm; flip to 0 if assembler rejects)
#define USE_ASYNC_LDS 1

typedef __attribute__((ext_vector_type(16))) _Float16 v16h;
typedef __attribute__((ext_vector_type(8)))  _Float16 v8h;
typedef __attribute__((ext_vector_type(8)))  float    v8f;
typedef __attribute__((ext_vector_type(4)))  float    v4f;

__device__ __forceinline__ v8f wmma_f16(v16h a, v16h b, v8f c) {
  return __builtin_amdgcn_wmma_f32_16x16x32_f16(false, a, false, b, (short)0, c,
                                                false, false);
}

__device__ __forceinline__ v8f zero8() {
  v8f z = {0.f, 0.f, 0.f, 0.f, 0.f, 0.f, 0.f, 0.f};
  return z;
}

#if USE_ASYNC_LDS
// Copy 16 bytes global -> LDS without a VGPR round trip (ASYNCcnt-tracked).
__device__ __forceinline__ void async_b128(void* lds, const void* g) {
  unsigned int  l   = (unsigned int)(unsigned long long)(uintptr_t)lds;  // LDS offset
  unsigned long long g64 = (unsigned long long)(uintptr_t)g;
  asm volatile("global_load_async_to_lds_b128 %0, %1, off"
               :: "v"(l), "v"(g64) : "memory");
}
__device__ __forceinline__ void wait_async0() {
  asm volatile("s_wait_asynccnt 0x0" ::: "memory");
}
#else
__device__ __forceinline__ void async_b128(void* lds, const void* g) {
  *(v8h*)lds = *(const v8h*)g;
}
__device__ __forceinline__ void wait_async0() {}
#endif

// Load one 16(M/N) x 32(K) f16 operand fragment from a row-major LDS tile.
// CDNA5 16-bit A/B layout: lanes 0-15 = rows 0-15 with K chunks {0..7,16..23},
// lanes 16-31 hold K chunks {8..15,24..31}.
__device__ __forceinline__ v16h load_frag(const _Float16* p, int row0, int k0, int ld) {
  const int lane = (int)(threadIdx.x & 31u);
  const _Float16* b = p + (row0 + (lane & 15)) * ld + k0 + ((lane >> 4) << 3);
  v8h lo = *(const v8h*)(b);
  v8h hi = *(const v8h*)(b + 16);
  v16h r;
#pragma unroll
  for (int i = 0; i < 8; ++i) { r[i] = lo[i]; r[i + 8] = hi[i]; }
  return r;
}

// ---------------------------------------------------------------------------
// One-time f32 -> f16 conversion (contiguous), 8 elems/thread.
// ---------------------------------------------------------------------------
__global__ __launch_bounds__(256) void cvt_f16_kernel(const float* __restrict__ in,
                                                      _Float16* __restrict__ out,
                                                      int n) {
  int i = (int)(blockIdx.x * 256u + threadIdx.x) * 8;
  if (i >= n) return;
  v4f f0 = *(const v4f*)(in + i);
  v4f f1 = *(const v4f*)(in + i + 4);
  v8h h;
#pragma unroll
  for (int j = 0; j < 4; ++j) { h[j] = (_Float16)f0[j]; h[4 + j] = (_Float16)f1[j]; }
  *(v8h*)(out + i) = h;
}

// ---------------------------------------------------------------------------
// One-time transpose + convert: out[n][k] = (f16) in[k][n].  32x32 LDS tiles,
// coalesced on both sides.
// ---------------------------------------------------------------------------
__global__ __launch_bounds__(256) void cvt_t_f16_kernel(const float* __restrict__ in,
                                                        _Float16* __restrict__ out,
                                                        int K, int N) {
  __shared__ float tile[32][33];
  const int kb = blockIdx.x * 32, nb = blockIdx.y * 32;
  const int tx = threadIdx.x & 31, ty = threadIdx.x >> 5;
#pragma unroll
  for (int r = ty; r < 32; r += 8)
    tile[r][tx] = in[(size_t)(kb + r) * N + nb + tx];
  __syncthreads();
#pragma unroll
  for (int r = ty; r < 32; r += 8)
    out[(size_t)(nb + r) * K + kb + tx] = (_Float16)tile[tx][r];
}

// ---------------------------------------------------------------------------
// C[M,N] = (A[M,K] @ W[K,N] + bias[N]) * scale, A and W^T pre-converted f16.
// 128x128 tile / 8 waves; K stepped by 32; stages are pure async b128 copies.
// F32OUT selects f32 (final projection) vs f16 (intermediate) output.
// ---------------------------------------------------------------------------
template <bool F32OUT>
__global__ __launch_bounds__(256) void gemm_bias_f16wmma(
    const _Float16* __restrict__ A,   // M x K row-major
    const _Float16* __restrict__ WT,  // N x K row-major (W transposed)
    const float* __restrict__ bias, void* __restrict__ Cv,
    int M, int N, int K, float scale)
{
  __shared__ alignas(16) _Float16 As[128][32];
  __shared__ alignas(16) _Float16 BsT[128][32];
  const int tid  = threadIdx.x;
  const int wave = tid >> 5;
  const int lane = tid & 31;
  const int row0 = blockIdx.x * 128;
  const int col0 = blockIdx.y * 128;
  const int wm = wave & 3;   // 4 groups of 32 rows
  const int wn = wave >> 2;  // 2 groups of 64 cols

  v8f acc[2][4];
#pragma unroll
  for (int i = 0; i < 2; ++i)
#pragma unroll
    for (int j = 0; j < 4; ++j) acc[i][j] = zero8();

  const int r  = tid >> 1;           // 0..127: tile row (A) / tile col (B)
  const int c0 = (tid & 1) * 16;     // 0 or 16 within K tile

  for (int k0 = 0; k0 < K; k0 += 32) {
    const _Float16* sa = A  + (size_t)(row0 + r) * K + k0 + c0;
    const _Float16* sb = WT + (size_t)(col0 + r) * K + k0 + c0;
    if (k0 + 32 < K) {
      __builtin_prefetch(sa + 32, 0, 0);   // next K tile
      __builtin_prefetch(sb + 32, 0, 0);
    }
    async_b128(&As[r][c0],      sa);
    async_b128(&As[r][c0 + 8],  sa + 8);
    async_b128(&BsT[r][c0],     sb);
    async_b128(&BsT[r][c0 + 8], sb + 8);
    wait_async0();
    __syncthreads();

    v16h a0 = load_frag(&As[0][0], wm * 32 + 0,  0, 32);
    v16h a1 = load_frag(&As[0][0], wm * 32 + 16, 0, 32);
#pragma unroll
    for (int nt = 0; nt < 4; ++nt) {
      v16h b = load_frag(&BsT[0][0], wn * 64 + nt * 16, 0, 32);
      acc[0][nt] = wmma_f16(a0, b, acc[0][nt]);
      acc[1][nt] = wmma_f16(a1, b, acc[1][nt]);
    }
    __syncthreads();
  }

#pragma unroll
  for (int mt = 0; mt < 2; ++mt)
#pragma unroll
    for (int nt = 0; nt < 4; ++nt) {
      int col  = col0 + wn * 64 + nt * 16 + (lane & 15);
      int rowb = row0 + wm * 32 + mt * 16 + ((lane >> 4) << 3);
      float bv = bias[col];
#pragma unroll
      for (int v = 0; v < 8; ++v) {
        float val = (acc[mt][nt][v] + bv) * scale;
        if (F32OUT) ((float*)Cv)[(size_t)(rowb + v) * N + col] = val;
        else        ((_Float16*)Cv)[(size_t)(rowb + v) * N + col] = (_Float16)val;
      }
    }
}

// ---------------------------------------------------------------------------
// phi: out[h,s,r] = 0.125 * exp( X_head[s,:] @ wg[:,r] - 0.5*||X_head[s,:]||^2 )
// One block (4 waves) per (head, 64-row block). 64x64x64 WMMA GEMM + exp.
// ---------------------------------------------------------------------------
__global__ __launch_bounds__(128) void phi_wmma(
    const _Float16* __restrict__ X,  // (S, DMODEL) f16, already *norm_d
    const float* __restrict__ wg,    // (DDEPTH, RKER) row-major f32
    _Float16* __restrict__ out)      // (NHEADS, S, RKER) f16
{
  __shared__ alignas(16) _Float16 Xs[64][64];
  __shared__ alignas(16) _Float16 WgT[64][64];  // [r][d]
  __shared__ float sumsq[64];
  const int tid  = threadIdx.x;
  const int wave = tid >> 5;
  const int lane = tid & 31;
  const int h  = blockIdx.x;
  const int s0 = blockIdx.y * 64;

  // stage X head-slice tile (pure f16 copies, async) + wg transposed (f32->f16)
  {
    int rr = tid >> 1, cc = (tid & 1) * 32;
    const _Float16* src = X + (size_t)(s0 + rr) * DMODEL + h * DDEPTH + cc;
#pragma unroll
    for (int j = 0; j < 32; j += 8) async_b128(&Xs[rr][cc + j], src + j);
  }
  {
    int rr = tid >> 1, d0 = (tid & 1) * 32;
#pragma unroll
    for (int dd = 0; dd < 32; ++dd)
      WgT[rr][d0 + dd] = (_Float16)wg[(size_t)(d0 + dd) * RKER + rr];
  }
  wait_async0();
  __syncthreads();

  if (tid < 64) {
    float ss = 0.f;
#pragma unroll
    for (int c = 0; c < 64; ++c) { float x = (float)Xs[tid][c]; ss += x * x; }
    sumsq[tid] = ss;
  }
  __syncthreads();

  const int m0 = wave * 16;
  v8f acc[4];
#pragma unroll
  for (int nt = 0; nt < 4; ++nt) acc[nt] = zero8();
#pragma unroll
  for (int ks = 0; ks < 2; ++ks) {
    v16h a = load_frag(&Xs[0][0], m0, ks * 32, 64);
#pragma unroll
    for (int nt = 0; nt < 4; ++nt)
      acc[nt] = wmma_f16(a, load_frag(&WgT[0][0], nt * 16, ks * 32, 64), acc[nt]);
  }

#pragma unroll
  for (int nt = 0; nt < 4; ++nt) {
    int col = nt * 16 + (lane & 15);
    int rl  = m0 + ((lane >> 4) << 3);
#pragma unroll
    for (int v = 0; v < 8; ++v) {
      float e = acc[nt][v] - 0.5f * sumsq[rl + v];
      out[((size_t)h * S_LEN + s0 + rl + v) * RKER + col] = (_Float16)(RSQR_D * __expf(e));
    }
  }
}

// ---------------------------------------------------------------------------
// Chunked causal linear-attention scan (FlashLinearAttention style).
// One block (4 waves) per head; 32-token chunks:
//   S      = Qc @ Kc^T                       (32x32, causal masked)
//   O      = Qc @ KV_prev + S_masked @ Vc    (32x64)
//   qk[t]  = rowsum(S_masked[t]) + Qc[t].Ksum_prev
//   KV^T  += Vc^T @ Kc     (f32 register accumulators, f16 LDS mirror)
// ---------------------------------------------------------------------------
__global__ __launch_bounds__(128) void linattn_scan_wmma(
    const _Float16* __restrict__ qp,  // (NHEADS, S, RKER) f16
    const _Float16* __restrict__ kp,  // (NHEADS, S, RKER) f16
    const _Float16* __restrict__ V,   // (S, DMODEL) f16
    _Float16* __restrict__ attn)      // (S, DMODEL) f16
{
  __shared__ alignas(16) _Float16 Qs[CHUNK][RKER];     // [t][r]
  __shared__ alignas(16) _Float16 Ks[CHUNK][RKER];     // [t][r]
  __shared__ alignas(16) _Float16 KsT[RKER][CHUNK];    // [r][t]
  __shared__ alignas(16) _Float16 VsT[DDEPTH][CHUNK];  // [d][t]
  __shared__ alignas(16) _Float16 KVT[DDEPTH][RKER];   // KV^T state f16 feed
  __shared__ alignas(16) _Float16 Sm[CHUNK][CHUNK];    // masked S
  __shared__ float Ksum[RKER];
  __shared__ float qk[CHUNK];
  __shared__ float ob[CHUNK][DDEPTH];

  const int tid  = threadIdx.x;
  const int wave = tid >> 5;
  const int lane = tid & 31;
  const int h = blockIdx.x;
  const size_t hS = (size_t)h * S_LEN;

  for (int i = tid; i < DDEPTH * RKER; i += 128) ((_Float16*)KVT)[i] = (_Float16)0.f;
  if (tid < RKER) Ksum[tid] = 0.f;
  v8f acc_kv[4];
#pragma unroll
  for (int nt = 0; nt < 4; ++nt) acc_kv[nt] = zero8();
  __syncthreads();

  for (int c = 0; c < S_LEN / CHUNK; ++c) {
    const int s0 = c * CHUNK;

    // phase 0: stage Qc, Kc (pure async copies) and Vc^T (strided gather)
    {
      int t = tid >> 2, r0 = (tid & 3) * 16;
      const _Float16* srcq = qp + (hS + s0 + t) * RKER + r0;
      const _Float16* srck = kp + (hS + s0 + t) * RKER + r0;
      async_b128(&Qs[t][r0],     srcq);
      async_b128(&Qs[t][r0 + 8], srcq + 8);
      async_b128(&Ks[t][r0],     srck);
      async_b128(&Ks[t][r0 + 8], srck + 8);
    }
    {
      int d = tid >> 1, t0 = (tid & 1) * 16;
      const _Float16* srcv = V + (size_t)(s0 + t0) * DMODEL + h * DDEPTH + d;
      v8h h0, h1;
#pragma unroll
      for (int i = 0; i < 8; ++i) h0[i] = srcv[(size_t)i * DMODEL];
#pragma unroll
      for (int i = 0; i < 8; ++i) h1[i] = srcv[(size_t)(i + 8) * DMODEL];
      *(v8h*)&VsT[d][t0] = h0; *(v8h*)&VsT[d][t0 + 8] = h1;
    }
    wait_async0();
    __syncthreads();

    // phase 1: KsT = Ks^T (LDS transpose)
    {
      int rr = tid >> 1, t0 = (tid & 1) * 16;
#pragma unroll
      for (int tt = 0; tt < 16; ++tt) KsT[rr][t0 + tt] = Ks[t0 + tt][rr];
    }
    __syncthreads();

    // phase 2: S = Qc @ Kc^T (B^T == Kc row-major), causal mask, store f16
    {
      const int mt = wave >> 1, nt = wave & 1;
      v8f s = zero8();
#pragma unroll
      for (int ks = 0; ks < 2; ++ks)
        s = wmma_f16(load_frag(&Qs[0][0], mt * 16, ks * 32, RKER),
                     load_frag(&Ks[0][0], nt * 16, ks * 32, RKER), s);
      int j  = nt * 16 + (lane & 15);
      int i0 = mt * 16 + ((lane >> 4) << 3);
#pragma unroll
      for (int v = 0; v < 8; ++v)
        Sm[i0 + v][j] = (_Float16)((j <= i0 + v) ? s[v] : 0.f);
    }
    __syncthreads();

    // phase 3a: wave0 computes normalizer qk[t] (uses pre-update Ksum)
    if (wave == 0) {
      int t = lane;
      float rs = 0.f;
#pragma unroll
      for (int j = 0; j < CHUNK; ++j) rs += (float)Sm[t][j];
      float dot = 0.f;
#pragma unroll
      for (int rr = 0; rr < RKER; ++rr) dot += (float)Qs[t][rr] * Ksum[rr];
      qk[t] = rs + dot;
    }

    // phase 3b: O = Qc @ KV_prev + Sm @ Vc ; wave w owns d-cols [w*16, w*16+16)
    {
      v8f o0 = zero8(), o1 = zero8();
      const int n0 = wave * 16;
#pragma unroll
      for (int ks = 0; ks < 2; ++ks) {
        v16h b = load_frag(&KVT[0][0], n0, ks * 32, RKER);  // B^T = KV^T [d][r]
        o0 = wmma_f16(load_frag(&Qs[0][0], 0,  ks * 32, RKER), b, o0);
        o1 = wmma_f16(load_frag(&Qs[0][0], 16, ks * 32, RKER), b, o1);
      }
      {
        v16h b = load_frag(&VsT[0][0], n0, 0, CHUNK);       // B^T = Vc^T [d][t]
        o0 = wmma_f16(load_frag(&Sm[0][0], 0,  0, CHUNK), b, o0);
        o1 = wmma_f16(load_frag(&Sm[0][0], 16, 0, CHUNK), b, o1);
      }
      int colD = n0 + (lane & 15);
      int r0 = (lane >> 4) << 3;
#pragma unroll
      for (int v = 0; v < 8; ++v) {
        ob[r0 + v][colD]      = o0[v];
        ob[16 + r0 + v][colD] = o1[v];
      }
    }

    // phase 4: KV^T += Vc^T @ Kc ; wave w owns d-rows [w*16, w*16+16)
#pragma unroll
    for (int nt = 0; nt < 4; ++nt)
      acc_kv[nt] = wmma_f16(load_frag(&VsT[0][0], wave * 16, 0, CHUNK),
                            load_frag(&KsT[0][0], nt * 16,   0, CHUNK), acc_kv[nt]);
    __syncthreads();

    // phase 5: write attn chunk (f16); refresh KVT; Ksum += colsum(Kc)
    {
      int t = tid >> 2, d0 = (tid & 3) * 16;
      float rcp = 1.f / (EPSV + qk[t]);
      _Float16* dst = attn + (size_t)(s0 + t) * DMODEL + h * DDEPTH + d0;
      v8h o0, o1;
#pragma unroll
      for (int j = 0; j < 8; ++j) {
        o0[j] = (_Float16)(ob[t][d0 + j] * rcp);
        o1[j] = (_Float16)(ob[t][d0 + 8 + j] * rcp);
      }
      *(v8h*)(dst)     = o0;
      *(v8h*)(dst + 8) = o1;
    }
#pragma unroll
    for (int nt = 0; nt < 4; ++nt) {
      int rr = nt * 16 + (lane & 15);
      int dd = wave * 16 + ((lane >> 4) << 3);
#pragma unroll
      for (int v = 0; v < 8; ++v) KVT[dd + v][rr] = (_Float16)acc_kv[nt][v];
    }
    if (tid < RKER) {
      float s = Ksum[tid];
#pragma unroll
      for (int t = 0; t < CHUNK; ++t) s += (float)KsT[tid][t];
      Ksum[tid] = s;
    }
    __syncthreads();
  }
}

// ---------------------------------------------------------------------------
extern "C" void kernel_launch(void* const* d_in, const int* in_sizes, int n_in,
                              void* d_out, int out_size, void* d_ws, size_t ws_size,
                              hipStream_t stream) {
  (void)in_sizes; (void)n_in; (void)out_size; (void)ws_size;
  const float* v    = (const float*)d_in[0];
  const float* k    = (const float*)d_in[1];
  const float* q    = (const float*)d_in[2];
  const float* wq_w = (const float*)d_in[3];
  const float* wq_b = (const float*)d_in[4];
  const float* wk_w = (const float*)d_in[5];
  const float* wk_b = (const float*)d_in[6];
  const float* wv_w = (const float*)d_in[7];
  const float* wv_b = (const float*)d_in[8];
  const float* wc_w = (const float*)d_in[9];
  const float* wc_b = (const float*)d_in[10];
  const float* wg   = (const float*)d_in[11];
  float* out = (float*)d_out;

  const size_t SM = (size_t)S_LEN * DMODEL;   // 2M elements
  const size_t WM = (size_t)DMODEL * DMODEL;  // 1M elements
  char* p = (char*)d_ws;
  auto alloc = [&](size_t bytes) { char* r = p; p += (bytes + 255) & ~(size_t)255; return r; };
  _Float16* q16  = (_Float16*)alloc(SM * 2);
  _Float16* k16  = (_Float16*)alloc(SM * 2);
  _Float16* v16  = (_Float16*)alloc(SM * 2);
  _Float16* wqT  = (_Float16*)alloc(WM * 2);
  _Float16* wkT  = (_Float16*)alloc(WM * 2);
  _Float16* wvT  = (_Float16*)alloc(WM * 2);
  _Float16* wcT  = (_Float16*)alloc(WM * 2);
  _Float16* Qp   = (_Float16*)alloc(SM * 2);  // projected q (scaled)
  _Float16* Kp   = (_Float16*)alloc(SM * 2);
  _Float16* Vp   = (_Float16*)alloc(SM * 2);
  _Float16* qpr  = (_Float16*)alloc(SM * 2);  // q_prime (NHEADS*S*RKER == SM)
  _Float16* kpr  = (_Float16*)alloc(SM * 2);
  _Float16* at16 = (_Float16*)alloc(SM * 2);  // pre-projection attention

  // one-time precision/layout conversion
  cvt_f16_kernel<<<(unsigned)(SM / 2048), 256, 0, stream>>>(q, q16, (int)SM);
  cvt_f16_kernel<<<(unsigned)(SM / 2048), 256, 0, stream>>>(k, k16, (int)SM);
  cvt_f16_kernel<<<(unsigned)(SM / 2048), 256, 0, stream>>>(v, v16, (int)SM);
  dim3 gt(DMODEL / 32, DMODEL / 32);
  cvt_t_f16_kernel<<<gt, 256, 0, stream>>>(wq_w, wqT, DMODEL, DMODEL);
  cvt_t_f16_kernel<<<gt, 256, 0, stream>>>(wk_w, wkT, DMODEL, DMODEL);
  cvt_t_f16_kernel<<<gt, 256, 0, stream>>>(wv_w, wvT, DMODEL, DMODEL);
  cvt_t_f16_kernel<<<gt, 256, 0, stream>>>(wc_w, wcT, DMODEL, DMODEL);

  dim3 gg(S_LEN / 128, DMODEL / 128);
  gemm_bias_f16wmma<false><<<gg, 256, 0, stream>>>(q16, wqT, wq_b, Qp, S_LEN, DMODEL, DMODEL, NORM_D);
  gemm_bias_f16wmma<false><<<gg, 256, 0, stream>>>(k16, wkT, wk_b, Kp, S_LEN, DMODEL, DMODEL, NORM_D);
  gemm_bias_f16wmma<false><<<gg, 256, 0, stream>>>(v16, wvT, wv_b, Vp, S_LEN, DMODEL, DMODEL, 1.0f);

  dim3 gp(NHEADS, S_LEN / 64);
  phi_wmma<<<gp, 128, 0, stream>>>(Qp, wg, qpr);
  phi_wmma<<<gp, 128, 0, stream>>>(Kp, wg, kpr);

  linattn_scan_wmma<<<NHEADS, 128, 0, stream>>>(qpr, kpr, Vp, at16);

  gemm_bias_f16wmma<true><<<gg, 256, 0, stream>>>(at16, wcT, wc_b, out, S_LEN, DMODEL, DMODEL, 1.0f);
}
